// EncoderDecoderGRU_29515015258716
// MI455X (gfx1250) — compile-verified, matrix operand order checked by reading, not run on hardware
//
#include <hip/hip_runtime.h>
#include <hip/hip_bf16.h>

typedef __attribute__((ext_vector_type(16))) __bf16 v16bf;
typedef __attribute__((ext_vector_type(8)))  __bf16 v8bf;
typedef __attribute__((ext_vector_type(8)))  float  v8f;

#define B_SZ   1024
#define T_IN   64
#define IN_SZ  8
#define HID    512
#define G3     1536
#define T_OUT  48

// ---------------------------------------------------------------------------
// helpers
// ---------------------------------------------------------------------------
__device__ __forceinline__ v16bf mk_frag(v8bf lo, v8bf hi) {
  v16bf r;
#pragma unroll
  for (int i = 0; i < 8; ++i) { r[i] = lo[i]; r[i + 8] = hi[i]; }
  return r;
}

__device__ __forceinline__ float sigmoidf_fast(float x) {
  return 1.0f / (1.0f + __expf(-x));
}

// CDNA5 async global->LDS copy, 16 bytes, tracked by ASYNCcnt.
// vdst = LDS byte offset (32-bit), vaddr = 64-bit global address, no SADDR.
__device__ __forceinline__ void async_copy_b128(unsigned lds_off, const void* gptr) {
  asm volatile("global_load_async_to_lds_b128 %0, %1, off"
               :: "v"(lds_off), "v"((unsigned long long)gptr)
               : "memory");
}
__device__ __forceinline__ unsigned lds_off_of(const void* p) {
  return (unsigned)(unsigned long long)p;   // low 32 bits = LDS allocation offset
}

// ---------------------------------------------------------------------------
// fp32 -> bf16 weight conversion
// ---------------------------------------------------------------------------
__global__ __launch_bounds__(256) void f2bf_kernel(const float* __restrict__ src,
                                                   __bf16* __restrict__ dst, int n) {
  int i = blockIdx.x * 256 + threadIdx.x;
  if (i < n) dst[i] = (__bf16)src[i];
}

// ---------------------------------------------------------------------------
// zero hidden state (fp32 + bf16 copies)
// ---------------------------------------------------------------------------
__global__ __launch_bounds__(256) void init_h_kernel(float* h0f, float* h1f,
                                                     __bf16* h0b, __bf16* h1b) {
  int i = blockIdx.x * 256 + threadIdx.x;
  if (i < B_SZ * HID) {
    h0f[i] = 0.0f; h1f[i] = 0.0f;
    h0b[i] = (__bf16)0.0f; h1b[i] = (__bf16)0.0f;
  }
}

// decoder seed input: clip(x[:, -1, :2], -5, 5)  (x pre-clip ±10 is subsumed)
__global__ __launch_bounds__(256) void init_decin_kernel(const float* __restrict__ x,
                                                         float* __restrict__ dec_inp) {
  int i = blockIdx.x * 256 + threadIdx.x;
  if (i < B_SZ * 2) {
    int b = i >> 1, j = i & 1;
    float v = x[b * (T_IN * IN_SZ) + (T_IN - 1) * IN_SZ + j];
    v = fminf(fmaxf(v, -5.0f), 5.0f);
    dec_inp[i] = v;
  }
}

// ---------------------------------------------------------------------------
// WMMA GEMM:  C[1024 x 1536] = A(1024 x K, bf16) @ W(1536 x K, bf16)^T + bias
// block tile 64(M) x 128(N), 256 threads = 8 waves arranged 2(M) x 4(N),
// each wave owns a 32x32 tile (2 A frags x 2 B frags -> 4 WMMAs, full reuse).
// Quad-buffered LDS tiles filled with global_load_async_to_lds_b128 and
// software-pipelined one K-tile ahead (one barrier per iteration; cheap &3
// buffer rotation; final iteration peeled so the loop body is branch-free).
// ---------------------------------------------------------------------------
__global__ __launch_bounds__(256) void gemm_bf16_kernel(const __bf16* __restrict__ A,
                                                        const __bf16* __restrict__ W,
                                                        const float* __restrict__ bias,
                                                        float* __restrict__ C, int K) {
  __shared__ __align__(16) __bf16 As[4][64][40];    // +8 bf16 pad per row
  __shared__ __align__(16) __bf16 Ws[4][128][40];

  const int tid   = threadIdx.x;
  const int wave  = tid >> 5;
  const int lane  = tid & 31;
  const int m_blk = blockIdx.x * 64;
  const int n_blk = blockIdx.y * 128;
  const int wm    = (wave & 1) * 32;     // wave M offset in tile (2 rows of waves)
  const int wn    = (wave >> 1) * 32;    // wave N offset in tile (4 cols of waves)
  const int lrow  = lane & 15;
  const int kb    = (lane >> 4) * 8;     // K sub-chunk base: 0 or 8 (ISA 16-bit layout)

  const int ar = tid >> 2, ac = (tid & 3) * 8;   // A tile 64x32: 1 x b128 per thread
  const int wr = tid >> 1, wc = (tid & 1) * 16;  // W tile 128x32: 2 x b128 per thread

  // per-thread async-copy issue of one K-tile into LDS buffer `buf` (3 ops)
  auto issue_tile = [&](int k0, int buf) {
    async_copy_b128(lds_off_of(&As[buf][ar][ac]),
                    &A[(size_t)(m_blk + ar) * K + k0 + ac]);
    async_copy_b128(lds_off_of(&Ws[buf][wr][wc]),
                    &W[(size_t)(n_blk + wr) * K + k0 + wc]);
    async_copy_b128(lds_off_of(&Ws[buf][wr][wc + 8]),
                    &W[(size_t)(n_blk + wr) * K + k0 + wc + 8]);
  };

  v8f acc00 = {}, acc01 = {}, acc10 = {}, acc11 = {};

  // 4 WMMAs on LDS buffer `cb` (2 A frags x 2 B frags, full operand reuse)
  auto compute_tile = [&](int cb) {
    // A fragments: lane<16 holds row M, K {0..7,16..23}; lane>=16 K {8..15,24..31}
    v16bf a0 = mk_frag(*(const v8bf*)&As[cb][wm +  0 + lrow][kb],
                       *(const v8bf*)&As[cb][wm +  0 + lrow][kb + 16]);
    v16bf a1 = mk_frag(*(const v8bf*)&As[cb][wm + 16 + lrow][kb],
                       *(const v8bf*)&As[cb][wm + 16 + lrow][kb + 16]);
    // B fragments (W^T): lane holds column n = lrow, same K striping
    v16bf b0 = mk_frag(*(const v8bf*)&Ws[cb][wn +  0 + lrow][kb],
                       *(const v8bf*)&Ws[cb][wn +  0 + lrow][kb + 16]);
    v16bf b1 = mk_frag(*(const v8bf*)&Ws[cb][wn + 16 + lrow][kb],
                       *(const v8bf*)&Ws[cb][wn + 16 + lrow][kb + 16]);
    acc00 = __builtin_amdgcn_wmma_f32_16x16x32_bf16(false, a0, false, b0, (short)0, acc00, false, false);
    acc01 = __builtin_amdgcn_wmma_f32_16x16x32_bf16(false, a0, false, b1, (short)0, acc01, false, false);
    acc10 = __builtin_amdgcn_wmma_f32_16x16x32_bf16(false, a1, false, b0, (short)0, acc10, false, false);
    acc11 = __builtin_amdgcn_wmma_f32_16x16x32_bf16(false, a1, false, b1, (short)0, acc11, false, false);
  };

  const int nk = K >> 5;          // K / 32
  issue_tile(0, 0);               // prologue: tile 0 in flight (3 ops)

  int kk = 0;
  for (; kk < nk - 1; ++kk) {
    // Pipeline: issue tile kk+1 into buffer (kk+1)&3. That buffer was last
    // READ at iteration kk-3; every wave passed the iteration kk-1 barrier
    // after retiring those ds_loads, so overwriting it now is safe.
    issue_tile((kk + 1) << 5, (kk + 1) & 3);
    // 6 outstanding; async loads complete in order -> <=3 means tile kk landed.
    asm volatile("s_wait_asynccnt 3" ::: "memory");
    __syncthreads();              // all waves' tile-kk async writes visible
    compute_tile(kk & 3);
  }
  // epilogue: last tile, nothing left to prefetch
  asm volatile("s_wait_asynccnt 0" ::: "memory");
  __syncthreads();
  compute_tile(kk & 3);

  // C/D layout: VGPR v, lane l -> M = v + (l>=16 ? 8:0), N = l%16
  const int mrow = m_blk + wm + ((lane >> 4) << 3);
#pragma unroll
  for (int s = 0; s < 4; ++s) {
    v8f a = (s == 0) ? acc00 : (s == 1) ? acc01 : (s == 2) ? acc10 : acc11;
    const int msub = (s >> 1) * 16;                 // acc0x -> M+0, acc1x -> M+16
    const int n    = n_blk + wn + (s & 1) * 16 + lrow;
    const float bv = bias[n];
#pragma unroll
    for (int v = 0; v < 8; ++v)
      C[(size_t)(mrow + msub + v) * G3 + n] = a[v] + bv;
  }
}

// ---------------------------------------------------------------------------
// GRU pointwise cell.  gi either from workspace buffer (layer 1) or computed
// inline from a small input (K = 8 encoder L0, K = 2 decoder L0).
// ---------------------------------------------------------------------------
__global__ __launch_bounds__(256) void gru_cell_kernel(
    const float* __restrict__ gi,        // null -> inline gi from xin
    const float* __restrict__ gh,
    const float* __restrict__ xin, int xdim, int xstride, float xclip,
    const float* __restrict__ Wih,       // fp32 (3H x xdim), used if gi==null
    const float* __restrict__ bih,
    const float* __restrict__ hprev,
    float* __restrict__ hout_f, __bf16* __restrict__ hout_b,
    float* __restrict__ hraw,            // optional unclipped copy (fc input)
    float hclip) {
  int idx = blockIdx.x * 256 + threadIdx.x;
  if (idx >= B_SZ * HID) return;
  int b = idx >> 9, k = idx & 511;

  float gir, giz, gin;
  if (gi) {
    const float* g = gi + (size_t)b * G3;
    gir = g[k]; giz = g[k + 512]; gin = g[k + 1024];
  } else {
    gir = bih[k]; giz = bih[k + 512]; gin = bih[k + 1024];
#pragma unroll 2
    for (int j = 0; j < xdim; ++j) {
      float xv = xin[(size_t)b * xstride + j];
      if (xclip > 0.0f) xv = fminf(fmaxf(xv, -xclip), xclip);
      gir += xv * Wih[k * xdim + j];
      giz += xv * Wih[(k + 512) * xdim + j];
      gin += xv * Wih[(k + 1024) * xdim + j];
    }
  }
  const float* g = gh + (size_t)b * G3;
  float ghr = g[k], ghz = g[k + 512], ghn = g[k + 1024];
  float h = hprev[idx];
  float r = sigmoidf_fast(gir + ghr);
  float z = sigmoidf_fast(giz + ghz);
  float n = tanhf(gin + r * ghn);
  float hn = (1.0f - z) * n + z * h;
  if (hraw) hraw[idx] = hn;                 // pre-clip value feeds the FC
  if (hclip > 0.0f) hn = fminf(fmaxf(hn, -hclip), hclip);
  hout_f[idx] = hn;
  hout_b[idx] = (__bf16)hn;
}

// ---------------------------------------------------------------------------
// decoder head: delta = tanh(h1_raw @ fcW^T + fcb); pred = clip(inp+delta,±5)
// one wave per batch row; 8 waves / block
// ---------------------------------------------------------------------------
__global__ __launch_bounds__(256) void dec_fc_kernel(const float* __restrict__ h1raw,
                                                     const float* __restrict__ fcW,
                                                     const float* __restrict__ fcb,
                                                     float* __restrict__ dec_inp,
                                                     float* __restrict__ out, int t) {
  int wave = threadIdx.x >> 5, lane = threadIdx.x & 31;
  int b = blockIdx.x * 8 + wave;
  const float* h = h1raw + (size_t)b * HID;
  float s0 = 0.0f, s1 = 0.0f;
#pragma unroll 4
  for (int k = lane; k < HID; k += 32) {
    float hv = h[k];
    s0 += hv * fcW[k];
    s1 += hv * fcW[HID + k];
  }
#pragma unroll
  for (int off = 16; off > 0; off >>= 1) {
    s0 += __shfl_down(s0, off, 32);
    s1 += __shfl_down(s1, off, 32);
  }
  if (lane == 0) {
    float p0 = fminf(fmaxf(dec_inp[b * 2 + 0] + tanhf(s0 + fcb[0]), -5.0f), 5.0f);
    float p1 = fminf(fmaxf(dec_inp[b * 2 + 1] + tanhf(s1 + fcb[1]), -5.0f), 5.0f);
    dec_inp[b * 2 + 0] = p0;
    dec_inp[b * 2 + 1] = p1;
    out[(size_t)b * (T_OUT * 2) + t * 2 + 0] = p0;
    out[(size_t)b * (T_OUT * 2) + t * 2 + 1] = p1;
  }
}

// ---------------------------------------------------------------------------
// host orchestration
// ---------------------------------------------------------------------------
extern "C" void kernel_launch(void* const* d_in, const int* in_sizes, int n_in,
                              void* d_out, int out_size, void* d_ws, size_t ws_size,
                              hipStream_t stream) {
  const float* x        = (const float*)d_in[0];
  const float* eWih0    = (const float*)d_in[1];
  const float* eWhh0    = (const float*)d_in[2];
  const float* ebih0    = (const float*)d_in[3];
  const float* ebhh0    = (const float*)d_in[4];
  const float* eWih1    = (const float*)d_in[5];
  const float* eWhh1    = (const float*)d_in[6];
  const float* ebih1    = (const float*)d_in[7];
  const float* ebhh1    = (const float*)d_in[8];
  const float* dWih0    = (const float*)d_in[9];
  const float* dWhh0    = (const float*)d_in[10];
  const float* dbih0    = (const float*)d_in[11];
  const float* dbhh0    = (const float*)d_in[12];
  const float* dWih1    = (const float*)d_in[13];
  const float* dWhh1    = (const float*)d_in[14];
  const float* dbih1    = (const float*)d_in[15];
  const float* dbhh1    = (const float*)d_in[16];
  const float* fcW      = (const float*)d_in[17];
  const float* fcb      = (const float*)d_in[18];
  float* out = (float*)d_out;

  // ---- workspace carve-out (all 256B aligned) ----
  char* ws = (char*)d_ws;
  size_t off = 0;
  auto alloc = [&](size_t bytes) -> void* {
    off = (off + 255) & ~(size_t)255;
    void* p = ws + off;
    off += bytes;
    return p;
  };
  const size_t HN  = (size_t)B_SZ * HID;
  const size_t WN  = (size_t)G3 * HID;
  float*  h0f = (float*)alloc(HN * 4);
  float*  h1f = (float*)alloc(HN * 4);
  __bf16* h0b = (__bf16*)alloc(HN * 2);
  __bf16* h1b = (__bf16*)alloc(HN * 2);
  float*  gi  = (float*)alloc((size_t)B_SZ * G3 * 4);
  float*  gh  = (float*)alloc((size_t)B_SZ * G3 * 4);
  float*  h1raw   = (float*)alloc(HN * 4);
  float*  dec_inp = (float*)alloc((size_t)B_SZ * 2 * 4);
  __bf16* eWhh0b = (__bf16*)alloc(WN * 2);
  __bf16* eWih1b = (__bf16*)alloc(WN * 2);
  __bf16* eWhh1b = (__bf16*)alloc(WN * 2);
  __bf16* dWhh0b = (__bf16*)alloc(WN * 2);
  __bf16* dWih1b = (__bf16*)alloc(WN * 2);
  __bf16* dWhh1b = (__bf16*)alloc(WN * 2);

  // ---- one-time per call: convert weights, init state ----
  {
    int n = (int)WN, g = (n + 255) / 256;
    f2bf_kernel<<<g, 256, 0, stream>>>(eWhh0, eWhh0b, n);
    f2bf_kernel<<<g, 256, 0, stream>>>(eWih1, eWih1b, n);
    f2bf_kernel<<<g, 256, 0, stream>>>(eWhh1, eWhh1b, n);
    f2bf_kernel<<<g, 256, 0, stream>>>(dWhh0, dWhh0b, n);
    f2bf_kernel<<<g, 256, 0, stream>>>(dWih1, dWih1b, n);
    f2bf_kernel<<<g, 256, 0, stream>>>(dWhh1, dWhh1b, n);
  }
  init_h_kernel<<<(int)(HN + 255) / 256, 256, 0, stream>>>(h0f, h1f, h0b, h1b);
  init_decin_kernel<<<(B_SZ * 2 + 255) / 256, 256, 0, stream>>>(x, dec_inp);

  const dim3 ggrid(B_SZ / 64, G3 / 128);          // 16 x 12
  const int  pgrid = (int)(HN / 256);             // 2048

  // ---- encoder: 64 steps ----
  for (int t = 0; t < T_IN; ++t) {
    const float hc = (t == T_IN - 1) ? 20.0f : 0.0f;  // clip final hidden only
    // layer 0: gh = h0 @ Whh0^T + bhh0 ; gi inline from x[:,t,:] (K=8, x clip ±10)
    gemm_bf16_kernel<<<ggrid, 256, 0, stream>>>(h0b, eWhh0b, ebhh0, gh, HID);
    gru_cell_kernel<<<pgrid, 256, 0, stream>>>(nullptr, gh,
        x + t * IN_SZ, IN_SZ, T_IN * IN_SZ, 10.0f, eWih0, ebih0,
        h0f, h0f, h0b, nullptr, hc);
    // layer 1: gi = h0 @ Wih1^T + bih1 ; gh = h1 @ Whh1^T + bhh1
    gemm_bf16_kernel<<<ggrid, 256, 0, stream>>>(h0b, eWih1b, ebih1, gi, HID);
    gemm_bf16_kernel<<<ggrid, 256, 0, stream>>>(h1b, eWhh1b, ebhh1, gh, HID);
    gru_cell_kernel<<<pgrid, 256, 0, stream>>>(gi, gh,
        nullptr, 0, 0, 0.0f, nullptr, nullptr,
        h1f, h1f, h1b, nullptr, hc);
  }

  // ---- decoder: 48 steps ----
  for (int t = 0; t < T_OUT; ++t) {
    // layer 0: gh = h0 @ dWhh0^T + bhh0 ; gi inline from dec_inp (K=2, pre-clipped)
    gemm_bf16_kernel<<<ggrid, 256, 0, stream>>>(h0b, dWhh0b, dbhh0, gh, HID);
    gru_cell_kernel<<<pgrid, 256, 0, stream>>>(nullptr, gh,
        dec_inp, 2, 2, 0.0f, dWih0, dbih0,
        h0f, h0f, h0b, nullptr, 20.0f);
    // layer 1
    gemm_bf16_kernel<<<ggrid, 256, 0, stream>>>(h0b, dWih1b, dbih1, gi, HID);
    gemm_bf16_kernel<<<ggrid, 256, 0, stream>>>(h1b, dWhh1b, dbhh1, gh, HID);
    gru_cell_kernel<<<pgrid, 256, 0, stream>>>(gi, gh,
        nullptr, 0, 0, 0.0f, nullptr, nullptr,
        h1f, h1f, h1b, h1raw, 20.0f);
    // head: delta = tanh(h1raw @ fcW^T + fcb); pred = clip(inp+delta, ±5)
    dec_fc_kernel<<<B_SZ / 8, 256, 0, stream>>>(h1raw, fcW, fcb, dec_inp, out, t);
  }
}
